// GPTModel_80719615361633
// MI455X (gfx1250) — matrix operand order repaired and checked by reading, not compiled, not run on hardware
//
#include <hip/hip_runtime.h>

// ---------------------------------------------------------------------------
// GPT-2-class forward for gfx1250 (MI455X): bf16 WMMA GEMMs, flash attention.
// ---------------------------------------------------------------------------

#define GV 50257
#define GD 768
#define GL 12
#define GH 12
#define GS 1024
#define GFF 3072
#define GB 2
#define GHD 64
#define LN_EPS 1e-5f

typedef __bf16 bf16_t;
typedef __attribute__((ext_vector_type(8)))  float  v8f;
typedef __attribute__((ext_vector_type(4)))  float  v4f;
typedef __attribute__((ext_vector_type(16))) __bf16 v16bf;
typedef __attribute__((ext_vector_type(8)))  __bf16 v8bf;
typedef __attribute__((ext_vector_type(4)))  __bf16 v4bf;

__device__ __forceinline__ v16bf pack16(const bf16_t* lo, const bf16_t* hi) {
  v8bf a = *(const v8bf*)lo;
  v8bf b = *(const v8bf*)hi;
  v16bf r;
#pragma unroll
  for (int i = 0; i < 8; ++i) { r[i] = a[i]; r[i + 8] = b[i]; }
  return r;
}

__device__ __forceinline__ v4bf cvt4(v4f f) {
  v4bf r;
#pragma unroll
  for (int i = 0; i < 4; ++i) r[i] = (bf16_t)f[i];
  return r;
}

// ---------------------------------------------------------------------------
// Embedding: x[b,s,:] = wte[id[b,s],:] + wpe[s,:]
// ---------------------------------------------------------------------------
__global__ __launch_bounds__(256) void embed_kernel(
    const int* __restrict__ ids, const float* __restrict__ wte,
    const float* __restrict__ wpe, float* __restrict__ x) {
  int idx = blockIdx.x * 256 + threadIdx.x;
  if (idx >= GB * GS * GD) return;
  int d = idx % GD;
  int bs = idx / GD;
  int s = bs % GS;
  int tok = ids[bs];
  x[idx] = wte[(size_t)tok * GD + d] + wpe[(size_t)s * GD + d];
}

// ---------------------------------------------------------------------------
// LayerNorm over D=768, one block (256 thr) per row; shfl-based reduction
// ---------------------------------------------------------------------------
__device__ __forceinline__ float block_sum256(float v, float* sbuf, int tid) {
#pragma unroll
  for (int d = 1; d < 32; d <<= 1) v += __shfl_xor(v, d, 32);
  if ((tid & 31) == 0) sbuf[tid >> 5] = v;
  __syncthreads();
  float t = 0.f;
#pragma unroll
  for (int w = 0; w < 8; ++w) t += sbuf[w];
  __syncthreads();
  return t;
}

__global__ __launch_bounds__(256) void ln_kernel(
    const float* __restrict__ x, const float* __restrict__ gamma,
    const float* __restrict__ beta, float* __restrict__ out) {
  __shared__ float sbuf[8];
  const int row = blockIdx.x;
  const int tid = threadIdx.x;
  const float* xr = x + (size_t)row * GD;

  float xv[3];
  float s = 0.f;
#pragma unroll
  for (int e = 0; e < 3; ++e) {
    xv[e] = xr[tid + e * 256];
    s += xv[e];
  }
  const float mean = block_sum256(s, sbuf, tid) * (1.0f / GD);

  float v = 0.f;
#pragma unroll
  for (int e = 0; e < 3; ++e) {
    float c = xv[e] - mean;
    v += c * c;
  }
  const float rstd = rsqrtf(block_sum256(v, sbuf, tid) * (1.0f / GD) + LN_EPS);

  float* outr = out + (size_t)row * GD;
#pragma unroll
  for (int e = 0; e < 3; ++e) {
    int d = tid + e * 256;
    outr[d] = (xv[e] - mean) * rstd * gamma[d] + beta[d];
  }
}

// ---------------------------------------------------------------------------
// Generic GEMM: C[M,N] = A[M,K] @ B + (bias) + (res)
//   A fp32 row-major, B fp32 ([K,N] or [N,K] if transB), fp32->bf16 in LDS,
//   fp32 accumulate via v_wmma_f32_16x16x32_bf16.
//   WG tile 64(M)x128(N), K-step 32, 8 waves (2x4), wave tile 32x32.
//   Register-pipelined: tile k+32 loads in flight while tile k does WMMA.
// ---------------------------------------------------------------------------
#define TM 64
#define TN 128
#define TK 32

__global__ __launch_bounds__(256) void gemm_kernel(
    const float* __restrict__ A, const float* __restrict__ Bw,
    const float* __restrict__ bias, const float* __restrict__ res,
    float* __restrict__ C, int M, int N, int K, int transB) {
  __shared__ __align__(16) bf16_t as[TM][TK];   // 4 KB
  __shared__ __align__(16) bf16_t bs[TN][TK];   // 8 KB (stored [n][k])

  const int tid = threadIdx.x;
  const int lane = tid & 31;
  const int wave = tid >> 5;
  const int wm = wave >> 2;   // 0..1
  const int wn = wave & 3;    // 0..3
  const int m0 = blockIdx.y * TM;
  const int n0 = blockIdx.x * TN;
  const int half = lane >> 4;
  const int l15 = lane & 15;

  v8f acc[2][2];
#pragma unroll
  for (int i = 0; i < 2; ++i)
#pragma unroll
    for (int j = 0; j < 2; ++j)
#pragma unroll
      for (int r = 0; r < 8; ++r) acc[i][j][r] = 0.0f;

  // --- A staging maps (no N dependence; M,K always tile-aligned here) ---
  // 64x32 floats = 512 float4 slots; 2 per thread.
  const int aRow0 = (tid + 0)   >> 3, aCol0 = ((tid + 0)   & 7) * 4;
  const int aRow1 = (tid + 256) >> 3, aCol1 = ((tid + 256) & 7) * 4;

  // --- fragment gather + 4 WMMAs on the staged tile ---
  auto mma_step = [&]() {
    v16bf af[2];
#pragma unroll
    for (int i = 0; i < 2; ++i) {
      const bf16_t* p = &as[wm * 32 + i * 16 + l15][half * 8];
      af[i] = pack16(p, p + 16);
    }
    v16bf bfrag[2];
#pragma unroll
    for (int j = 0; j < 2; ++j) {
      const bf16_t* p = &bs[wn * 32 + j * 16 + l15][half * 8];
      bfrag[j] = pack16(p, p + 16);
    }
#pragma unroll
    for (int i = 0; i < 2; ++i)
#pragma unroll
      for (int j = 0; j < 2; ++j)
        acc[i][j] = __builtin_amdgcn_wmma_f32_16x16x32_bf16(
            false, af[i], false, bfrag[j], (short)0, acc[i][j], false, false);
  };

  const bool full = (n0 + TN <= N);

  if (full) {
    // ---- fast path: vectorized, register-pipelined ----
    v4f aR[2];
    float bR[16];

    auto loadA = [&](int k0) {
      aR[0] = *(const v4f*)&A[(size_t)(m0 + aRow0) * K + k0 + aCol0];
      aR[1] = *(const v4f*)&A[(size_t)(m0 + aRow1) * K + k0 + aCol1];
    };
    auto storeA = [&]() {
      *(v4bf*)&as[aRow0][aCol0] = cvt4(aR[0]);
      *(v4bf*)&as[aRow1][aCol1] = cvt4(aR[1]);
    };
    auto loadB = [&](int k0) {
      if (!transB) {
        // tasks: (nn 0..127) x (kg 0..7); 4 strided b32 loads each,
        // coalesced across threads along n.
#pragma unroll
        for (int e = 0; e < 4; ++e) {
          int task = tid + e * 256;
          int nn = task & 127, kg = task >> 7;
#pragma unroll
          for (int j = 0; j < 4; ++j)
            bR[e * 4 + j] = Bw[(size_t)(k0 + kg * 4 + j) * N + (n0 + nn)];
        }
      } else {
        // B stored [N,K]: contiguous float4 along K.
#pragma unroll
        for (int e = 0; e < 4; ++e) {
          int task = tid + e * 256;
          int nn = task >> 3, kg = task & 7;
          v4f q = *(const v4f*)&Bw[(size_t)(n0 + nn) * K + k0 + kg * 4];
#pragma unroll
          for (int j = 0; j < 4; ++j) bR[e * 4 + j] = q[j];
        }
      }
    };
    auto storeB = [&]() {
#pragma unroll
      for (int e = 0; e < 4; ++e) {
        int task = tid + e * 256;
        int nn = transB ? (task >> 3) : (task & 127);
        int kg = transB ? (task & 7) : (task >> 7);
        v4f q;
#pragma unroll
        for (int j = 0; j < 4; ++j) q[j] = bR[e * 4 + j];
        *(v4bf*)&bs[nn][kg * 4] = cvt4(q);
      }
    };

    loadA(0);
    loadB(0);
    int k0 = 0;
    while (true) {
      storeA();
      storeB();
      __syncthreads();
      const bool more = (k0 + TK < K);
      if (more) {
        loadA(k0 + TK);
        loadB(k0 + TK);
        if (k0 + 2 * TK < K) {  // keep L2 one more tile ahead
          __builtin_prefetch(&A[(size_t)(m0 + aRow0) * K + k0 + 2 * TK], 0, 2);
          if (!transB)
            __builtin_prefetch(&Bw[(size_t)(k0 + 2 * TK) * N + n0 + (tid & 127)], 0, 2);
          else
            __builtin_prefetch(&Bw[(size_t)(n0 + (tid >> 1)) * K + k0 + 2 * TK], 0, 2);
        }
      }
      mma_step();
      __syncthreads();
      if (!more) break;
      k0 += TK;
    }
  } else {
    // ---- edge path (ragged N, lm_head last column only): guarded scalar ----
    for (int k0 = 0; k0 < K; k0 += TK) {
      {  // A: vectorized (always in-bounds)
        v4f a0 = *(const v4f*)&A[(size_t)(m0 + aRow0) * K + k0 + aCol0];
        v4f a1 = *(const v4f*)&A[(size_t)(m0 + aRow1) * K + k0 + aCol1];
        *(v4bf*)&as[aRow0][aCol0] = cvt4(a0);
        *(v4bf*)&as[aRow1][aCol1] = cvt4(a1);
      }
      if (!transB) {
#pragma unroll
        for (int e = 0; e < 16; ++e) {
          int idx = tid + e * 256;
          int kk = idx >> 7, nn = idx & 127;
          int n = n0 + nn;
          float v = (n < N) ? Bw[(size_t)(k0 + kk) * N + n] : 0.0f;
          bs[nn][kk] = (bf16_t)v;
        }
      } else {
#pragma unroll
        for (int e = 0; e < 16; ++e) {
          int idx = tid + e * 256;
          int nn = idx >> 5, kk = idx & 31;
          int n = n0 + nn;
          float v = (n < N) ? Bw[(size_t)n * K + (k0 + kk)] : 0.0f;
          bs[nn][kk] = (bf16_t)v;
        }
      }
      __syncthreads();
      mma_step();
      __syncthreads();
    }
  }

  // ---- epilogue: bias + residual + store ----
#pragma unroll
  for (int i = 0; i < 2; ++i)
#pragma unroll
    for (int j = 0; j < 2; ++j) {
      int col = n0 + wn * 32 + j * 16 + l15;
      if (col < N) {
        float bc = bias ? bias[col] : 0.0f;
#pragma unroll
        for (int r = 0; r < 8; ++r) {
          int row = m0 + wm * 32 + i * 16 + half * 8 + r;
          size_t idx = (size_t)row * N + col;
          float v = acc[i][j][r] + bc;
          if (res) v += res[idx];
          C[idx] = v;
        }
      }
    }
}

// ---------------------------------------------------------------------------
// Split QKV fp32 [B,S,3D] -> per-head bf16: q,k row-major [BH,S,64]; v^T [BH,64,S]
// ---------------------------------------------------------------------------
__global__ __launch_bounds__(256) void split_qkv_kernel(
    const float* __restrict__ qkv, bf16_t* __restrict__ qb,
    bf16_t* __restrict__ kb, bf16_t* __restrict__ vtb) {
  int idx = blockIdx.x * 256 + threadIdx.x;
  if (idx >= GB * GS * GD) return;
  int d = idx % GHD;
  int h = (idx / GHD) % GH;
  int s = (idx / (GHD * GH)) % GS;
  int b = idx / (GHD * GH * GS);
  size_t row = (size_t)(b * GS + s) * (3 * GD) + h * GHD + d;
  int bh = b * GH + h;
  qb[((size_t)bh * GS + s) * GHD + d]  = (bf16_t)qkv[row];
  kb[((size_t)bh * GS + s) * GHD + d]  = (bf16_t)qkv[row + GD];
  vtb[((size_t)bh * GHD + d) * GS + s] = (bf16_t)qkv[row + 2 * GD];
}

// ---------------------------------------------------------------------------
// Flash attention (causal). Grid: (S/64, B*H). Block: 128 (4 waves).
// Each wave owns 16 query rows; streams 64-key tiles with online softmax.
// q@k^T and P@v both via v_wmma_f32_16x16x32_bf16.
// ---------------------------------------------------------------------------
__global__ __launch_bounds__(128) void attn_kernel(
    const bf16_t* __restrict__ qb, const bf16_t* __restrict__ kb,
    const bf16_t* __restrict__ vtb, float* __restrict__ aout) {
  const int qt = blockIdx.x;
  const int bh = blockIdx.y;
  const int b = bh / GH;
  const int h = bh % GH;
  const bf16_t* q  = qb  + (size_t)bh * GS * GHD;
  const bf16_t* kk = kb  + (size_t)bh * GS * GHD;
  const bf16_t* vt = vtb + (size_t)bh * GHD * GS;

  __shared__ __align__(16) bf16_t p_lds[4][16][64];  // per-wave P tile, 8 KB

  const int tid = threadIdx.x;
  const int lane = tid & 31;
  const int wave = tid >> 5;
  const int half = lane >> 4;
  const int l15 = lane & 15;
  const int qrow0 = qt * 64 + wave * 16;
  const float scale = 0.125f;  // 1/sqrt(64)

  v8f O[4];
  float rowM[8], rowL[8];
#pragma unroll
  for (int t = 0; t < 4; ++t)
#pragma unroll
    for (int r = 0; r < 8; ++r) O[t][r] = 0.0f;
#pragma unroll
  for (int r = 0; r < 8; ++r) { rowM[r] = -3.0e38f; rowL[r] = 0.0f; }

  for (int kt = 0; kt <= qt; ++kt) {
    // ---- scores S = q @ k^T over d=64 (2 K-steps of 32) ----
    v8f Sacc[4];
#pragma unroll
    for (int t = 0; t < 4; ++t)
#pragma unroll
      for (int r = 0; r < 8; ++r) Sacc[t][r] = 0.0f;

#pragma unroll
    for (int ks = 0; ks < 2; ++ks) {
      const bf16_t* pq = q + (size_t)(qrow0 + l15) * GHD + ks * 32 + half * 8;
      v16bf aq = pack16(pq, pq + 16);
#pragma unroll
      for (int t = 0; t < 4; ++t) {
        const bf16_t* pk =
            kk + (size_t)(kt * 64 + t * 16 + l15) * GHD + ks * 32 + half * 8;
        v16bf bk = pack16(pk, pk + 16);
        Sacc[t] = __builtin_amdgcn_wmma_f32_16x16x32_bf16(
            false, aq, false, bk, (short)0, Sacc[t], false, false);
      }
    }

    // ---- scale, causal mask, online softmax stats ----
    float alpha[8];
#pragma unroll
    for (int r = 0; r < 8; ++r) {
      int m = qrow0 + half * 8 + r;
      float mx = -3.0e38f;
#pragma unroll
      for (int t = 0; t < 4; ++t) {
        int key = kt * 64 + t * 16 + l15;
        float v = Sacc[t][r] * scale;
        if (key > m) v = -1.0e30f;
        Sacc[t][r] = v;
        mx = fmaxf(mx, v);
      }
#pragma unroll
      for (int d = 1; d < 16; d <<= 1) mx = fmaxf(mx, __shfl_xor(mx, d, 32));
      float nm = fmaxf(rowM[r], mx);
      alpha[r] = __expf(rowM[r] - nm);
      rowM[r] = nm;
      float sum = 0.0f;
#pragma unroll
      for (int t = 0; t < 4; ++t) {
        float p = __expf(Sacc[t][r] - nm);
        Sacc[t][r] = p;
        sum += p;
      }
#pragma unroll
      for (int d = 1; d < 16; d <<= 1) sum += __shfl_xor(sum, d, 32);
      rowL[r] = rowL[r] * alpha[r] + sum;
    }

    // ---- P to LDS (accumulator layout -> A-fragment layout) ----
#pragma unroll
    for (int t = 0; t < 4; ++t)
#pragma unroll
      for (int r = 0; r < 8; ++r)
        p_lds[wave][half * 8 + r][t * 16 + l15] = (bf16_t)Sacc[t][r];

    // ---- O = alpha*O + P @ v ----
#pragma unroll
    for (int t = 0; t < 4; ++t)
#pragma unroll
      for (int r = 0; r < 8; ++r) O[t][r] *= alpha[r];

#pragma unroll
    for (int ks = 0; ks < 2; ++ks) {
      const bf16_t* pp = &p_lds[wave][l15][ks * 32 + half * 8];
      v16bf ap = pack16(pp, pp + 16);
#pragma unroll
      for (int t = 0; t < 4; ++t) {
        const bf16_t* pv =
            vt + (size_t)(t * 16 + l15) * GS + kt * 64 + ks * 32 + half * 8;
        v16bf bv = pack16(pv, pv + 16);
        O[t] = __builtin_amdgcn_wmma_f32_16x16x32_bf16(
            false, ap, false, bv, (short)0, O[t], false, false);
      }
    }
  }

  // ---- normalize and scatter into [B,S,D] ----
#pragma unroll
  for (int t = 0; t < 4; ++t) {
    int dcol = t * 16 + l15;
#pragma unroll
    for (int r = 0; r < 8; ++r) {
      int m = qrow0 + half * 8 + r;
      aout[(size_t)(b * GS + m) * GD + h * GHD + dcol] = O[t][r] / rowL[r];
    }
  }
}

// ---------------------------------------------------------------------------
// SwiGLU elementwise: gv = silu(g) * v
// ---------------------------------------------------------------------------
__global__ __launch_bounds__(256) void silu_mul_kernel(
    const float* __restrict__ g, const float* __restrict__ v,
    float* __restrict__ gv, int n) {
  int idx = blockIdx.x * 256 + threadIdx.x;
  if (idx >= n) return;
  float x = g[idx];
  gv[idx] = (x / (1.0f + __expf(-x))) * v[idx];
}

// ---------------------------------------------------------------------------
// Host-side orchestration
// ---------------------------------------------------------------------------
extern "C" void kernel_launch(void* const* d_in, const int* in_sizes, int n_in,
                              void* d_out, int out_size, void* d_ws,
                              size_t ws_size, hipStream_t stream) {
  (void)in_sizes; (void)n_in; (void)out_size; (void)ws_size;
  const int*   ids   = (const int*)d_in[0];
  const float* wte   = (const float*)d_in[1];
  const float* wpe   = (const float*)d_in[2];
  const float* ln1_s = (const float*)d_in[3];
  const float* ln1_b = (const float*)d_in[4];
  const float* w_qkv = (const float*)d_in[5];
  const float* b_qkv = (const float*)d_in[6];
  const float* w_out = (const float*)d_in[7];
  const float* b_out = (const float*)d_in[8];
  const float* ln2_s = (const float*)d_in[9];
  const float* ln2_b = (const float*)d_in[10];
  const float* w_fc1 = (const float*)d_in[11];
  const float* b_fc1 = (const float*)d_in[12];
  const float* w_fc2 = (const float*)d_in[13];
  const float* b_fc2 = (const float*)d_in[14];
  const float* w_fc3 = (const float*)d_in[15];
  const float* b_fc3 = (const float*)d_in[16];
  const float* lnf_s = (const float*)d_in[17];
  const float* lnf_b = (const float*)d_in[18];

  const int M = GB * GS;  // 2048 token rows

  // workspace carve-up (all chunks are multiples of 256 bytes)
  char* ws = (char*)d_ws;
  float*  x    = (float*)ws;                 ws += (size_t)M * GD * 4;
  float*  hbuf = (float*)ws;                 ws += (size_t)M * GD * 4;
  float*  qkv  = (float*)ws;                 ws += (size_t)M * 3 * GD * 4;
  bf16_t* qb   = (bf16_t*)ws;                ws += (size_t)GB * GH * GS * GHD * 2;
  bf16_t* kb   = (bf16_t*)ws;                ws += (size_t)GB * GH * GS * GHD * 2;
  bf16_t* vtb  = (bf16_t*)ws;                ws += (size_t)GB * GH * GHD * GS * 2;
  float*  aout = (float*)ws;                 ws += (size_t)M * GD * 4;
  float*  gbuf = (float*)ws;                 ws += (size_t)M * GFF * 4;
  float*  vbuf = (float*)ws;                 ws += (size_t)M * GFF * 4;
  float*  gv   = (float*)ws;                 ws += (size_t)M * GFF * 4;

  const int nThreads = 256;
  const int nEmbed = (GB * GS * GD + nThreads - 1) / nThreads;

  embed_kernel<<<nEmbed, nThreads, 0, stream>>>(ids, wte, wpe, x);

  for (int l = 0; l < GL; ++l) {
    // --- attention block ---
    ln_kernel<<<M, 256, 0, stream>>>(x, ln1_s + (size_t)l * GD,
                                     ln1_b + (size_t)l * GD, hbuf);
    gemm_kernel<<<dim3((3 * GD + TN - 1) / TN, M / TM), 256, 0, stream>>>(
        hbuf, w_qkv + (size_t)l * GD * 3 * GD, b_qkv + (size_t)l * 3 * GD,
        nullptr, qkv, M, 3 * GD, GD, 0);
    split_qkv_kernel<<<nEmbed, nThreads, 0, stream>>>(qkv, qb, kb, vtb);
    attn_kernel<<<dim3(GS / 64, GB * GH), 128, 0, stream>>>(qb, kb, vtb, aout);
    gemm_kernel<<<dim3((GD + TN - 1) / TN, M / TM), 256, 0, stream>>>(
        aout, w_out + (size_t)l * GD * GD, b_out + (size_t)l * GD,
        x, x, M, GD, GD, 0);  // x += attn_proj

    // --- SwiGLU MLP block ---
    ln_kernel<<<M, 256, 0, stream>>>(x, ln2_s + (size_t)l * GD,
                                     ln2_b + (size_t)l * GD, hbuf);
    gemm_kernel<<<dim3((GFF + TN - 1) / TN, M / TM), 256, 0, stream>>>(
        hbuf, w_fc1 + (size_t)l * GD * GFF, b_fc1 + (size_t)l * GFF,
        nullptr, gbuf, M, GFF, GD, 0);
    gemm_kernel<<<dim3((GFF + TN - 1) / TN, M / TM), 256, 0, stream>>>(
        hbuf, w_fc2 + (size_t)l * GD * GFF, b_fc2 + (size_t)l * GFF,
        nullptr, vbuf, M, GFF, GD, 0);
    silu_mul_kernel<<<(M * GFF + nThreads - 1) / nThreads, nThreads, 0,
                      stream>>>(gbuf, vbuf, gv, M * GFF);
    gemm_kernel<<<dim3((GD + TN - 1) / TN, M / TM), 256, 0, stream>>>(
        gv, w_fc3 + (size_t)l * GFF * GD, b_fc3 + (size_t)l * GD,
        x, x, M, GD, GFF, 0);  // x += mlp
  }

  // --- final LN + tied lm_head (B = wte stored [V,D] -> transB path) ---
  ln_kernel<<<M, 256, 0, stream>>>(x, lnf_s, lnf_b, hbuf);
  gemm_kernel<<<dim3((GV + TN - 1) / TN, M / TM), 256, 0, stream>>>(
      hbuf, wte, nullptr, nullptr, (float*)d_out, M, GV, GD, 1);
}
